// PatchedMatmul_5781025980792
// MI455X (gfx1250) — 8-bit matrix operand order repaired and checked by reading, compiled, not run on hardware
//
#include <hip/hip_runtime.h>


typedef int          v16i __attribute__((ext_vector_type(16)));
typedef float        v8f  __attribute__((ext_vector_type(8)));
typedef unsigned int v4u  __attribute__((ext_vector_type(4)));

#define S_DIM 2048
#define T_DIM 2048
#define D_DIM 128
#define LDSS  144
#define OSTR  144

static __device__ __forceinline__ unsigned short f32_to_bf16_rne(float f) {
    unsigned int u = __float_as_uint(f);
    u += 0x7fffu + ((u >> 16) & 1u);
    return (unsigned short)(u >> 16);
}

__global__ __launch_bounds__(256) void fp8_gemm_bf16_kernel(
    const float* __restrict__ A,
    const float* __restrict__ Bm,
    const float* __restrict__ scale_a,
    const float* __restrict__ scale_b,
    unsigned short* __restrict__ out)
{
    __shared__ unsigned char lds[2 * 128 * LDSS];
    unsigned char* a_lds = lds;
    unsigned char* b_lds = lds + 128 * LDSS;
    unsigned short* stage = (unsigned short*)lds;

    const int tid = threadIdx.x;
    const int bh  = blockIdx.z;
    const int s0  = blockIdx.y * 128;
    const int t0  = blockIdx.x * 128;

    const float sa      = scale_a[0];
    const float sb      = scale_b[0];
    const float inv_sa  = 1.0f / sa;
    const float inv_sb  = 1.0f / sb;
    const float dequant = sa * sb;

    {
        const float* a_base = A + ((size_t)bh * S_DIM + s0) * D_DIM;
        #pragma unroll
        for (int i = 0; i < 16; ++i) {
            const int f   = i * 256 + tid;
            const int row = f >> 5;
            const int c4  = (f & 31) * 4;
            const float4 v = *(const float4*)(a_base + (size_t)row * D_DIM + c4);
            const float x0 = fminf(fmaxf(v.x * inv_sa, -448.f), 448.f);
            const float x1 = fminf(fmaxf(v.y * inv_sa, -448.f), 448.f);
            const float x2 = fminf(fmaxf(v.z * inv_sa, -448.f), 448.f);
            const float x3 = fminf(fmaxf(v.w * inv_sa, -448.f), 448.f);
            int packed = __builtin_amdgcn_cvt_pk_fp8_f32(x0, x1, 0, false);
            packed     = __builtin_amdgcn_cvt_pk_fp8_f32(x2, x3, packed, true);
            *(unsigned int*)(a_lds + row * LDSS + c4) = (unsigned int)packed;
        }
    }

    {
        const float* b_base = Bm + (size_t)bh * D_DIM * T_DIM + t0;
        #pragma unroll
        for (int i = 0; i < 16; ++i) {
            const int f  = i * 256 + tid;
            const int d  = f >> 5;
            const int t4 = (f & 31) * 4;
            const float4 v = *(const float4*)(b_base + (size_t)d * T_DIM + t4);
            const float x0 = fminf(fmaxf(v.x * inv_sb, -448.f), 448.f);
            const float x1 = fminf(fmaxf(v.y * inv_sb, -448.f), 448.f);
            const float x2 = fminf(fmaxf(v.z * inv_sb, -448.f), 448.f);
            const float x3 = fminf(fmaxf(v.w * inv_sb, -448.f), 448.f);
            int packed = __builtin_amdgcn_cvt_pk_fp8_f32(x0, x1, 0, false);
            packed     = __builtin_amdgcn_cvt_pk_fp8_f32(x2, x3, packed, true);
            b_lds[(t4 + 0) * LDSS + d] = (unsigned char)( packed        & 0xff);
            b_lds[(t4 + 1) * LDSS + d] = (unsigned char)((packed >>  8) & 0xff);
            b_lds[(t4 + 2) * LDSS + d] = (unsigned char)((packed >> 16) & 0xff);
            b_lds[(t4 + 3) * LDSS + d] = (unsigned char)((packed >> 24) & 0xff);
        }
    }

    __syncthreads();

    const int lane = tid & 31;
    const int w    = tid >> 5;
    const int wm   = w >> 2;
    const int wn   = w & 3;
    const int lm   = lane & 15;
    const int hi   = lane >> 4;

    v16i afrag[4];
    #pragma unroll
    for (int mi = 0; mi < 4; ++mi) {
        const int m = wm * 64 + mi * 16 + lm;
        const unsigned char* arow = a_lds + m * LDSS + hi * 16;
        #pragma unroll
        for (int p = 0; p < 8; ++p) {
            const int k = ((p >> 1) << 5) + ((p & 1) << 3);
            const uint2 q = *(const uint2*)(arow + k);
            afrag[mi][2 * p + 0] = (int)q.x;
            afrag[mi][2 * p + 1] = (int)q.y;
        }
    }

    v16i bfrag[2];
    #pragma unroll
    for (int ni = 0; ni < 2; ++ni) {
        const int n = wn * 32 + ni * 16 + lm;
        const unsigned char* bcol = b_lds + n * LDSS + hi * 16;
        #pragma unroll
        for (int g = 0; g < 4; ++g) {
            const uint4 q = *(const uint4*)(bcol + g * 32);
            bfrag[ni][4 * g + 0] = (int)q.x;
            bfrag[ni][4 * g + 1] = (int)q.y;
            bfrag[ni][4 * g + 2] = (int)q.z;
            bfrag[ni][4 * g + 3] = (int)q.w;
        }
    }

    v8f acc[4][2];
    #pragma unroll
    for (int mi = 0; mi < 4; ++mi)
        #pragma unroll
        for (int ni = 0; ni < 2; ++ni)
            acc[mi][ni] = (v8f)(0.0f);

    #pragma unroll
    for (int mi = 0; mi < 4; ++mi) {
        #pragma unroll
        for (int ni = 0; ni < 2; ++ni) {
            acc[mi][ni] = __builtin_amdgcn_wmma_f32_16x16x128_fp8_fp8(
                afrag[mi], bfrag[ni], (short)0, acc[mi][ni],
                 false,  false);
        }
    }

    __syncthreads();

    #pragma unroll
    for (int mi = 0; mi < 4; ++mi) {
        #pragma unroll
        for (int ni = 0; ni < 2; ++ni) {
            const int colL = wn * 32 + ni * 16 + lm;
            #pragma unroll
            for (int r = 0; r < 8; ++r) {
                const int rowL = wm * 64 + mi * 16 + hi * 8 + r;
                stage[rowL * OSTR + colL] =
                    f32_to_bf16_rne(acc[mi][ni][r] * dequant);
            }
        }
    }

    __syncthreads();

    unsigned short* obase = out + (size_t)bh * S_DIM * T_DIM;
    #pragma unroll
    for (int i = 0; i < 8; ++i) {
        const int c    = i * 256 + tid;
        const int row  = c >> 4;
        const int off8 = (c & 15) * 8;
        const v4u q = *(const v4u*)(stage + row * OSTR + off8);
        __builtin_nontemporal_store(
            q, (v4u*)(obase + (size_t)(s0 + row) * T_DIM + t0 + off8));
    }
}

extern "C" void kernel_launch(void* const* d_in, const int* in_sizes, int n_in,
                              void* d_out, int out_size, void* d_ws, size_t ws_size,
                              hipStream_t stream) {
    (void)in_sizes; (void)n_in; (void)d_ws; (void)ws_size; (void)out_size;
    const float* input       = (const float*)d_in[0];
    const float* other       = (const float*)d_in[1];
    const float* scale_input = (const float*)d_in[2];
    const float* scale_other = (const float*)d_in[3];
    unsigned short* out = (unsigned short*)d_out;

    dim3 grid(T_DIM / 128, S_DIM / 128, 64);
    dim3 block(256);
    fp8_gemm_bf16_kernel<<<grid, block, 0, stream>>>(input, other, scale_input,
                                                     scale_other, out);
}
